// GAT_13735305413408
// MI455X (gfx1250) — compile-verified, matrix operand order checked by reading, not run on hardware
//
#include <hip/hip_runtime.h>
#include <math.h>

typedef __attribute__((ext_vector_type(16))) _Float16 v16h;
typedef __attribute__((ext_vector_type(8)))  float    v8f;

#define NEG_SLOPE 0.2f

// ---------------------------------------------------------------------------
// Pack W (row-major K=128 x D fp32) into per-lane WMMA B-fragment layout, f16.
// Fragment f = nt*4 + kk  (nt = N-tile, kk = K-step). Each of 32 lanes owns 16
// halfs (32 B): j<8 -> W[ka+j][col], j>=8 -> W[ka+8+j][col], where
// ka = kk*32 + (lane>=16 ? 8 : 0), col = nt*16 + (lane%16).
// ---------------------------------------------------------------------------
__global__ void gat_pack_w(const float* __restrict__ W, _Float16* __restrict__ Wp,
                           int D) {
  int t = blockIdx.x * blockDim.x + threadIdx.x;
  int nfrag = (D / 16) * 4;
  if (t >= nfrag * 32) return;
  int f = t >> 5, lane = t & 31;
  int nt = f >> 2, kk = f & 3;
  bool hi = lane >= 16;
  int r = hi ? lane - 16 : lane;
  int bcol = nt * 16 + r;
  int ka = kk * 32 + (hi ? 8 : 0);
  _Float16* out = Wp + (size_t)t * 16;
  #pragma unroll
  for (int j = 0; j < 8; ++j) {
    out[j]     = (_Float16)W[(ka + j)      * D + bcol];
    out[j + 8] = (_Float16)W[(ka + 16 + j) * D + bcol];
  }
}

// ---------------------------------------------------------------------------
// fp32 -> f16 feature conversion (layer-1 input only; n % 4 == 0)
// ---------------------------------------------------------------------------
__global__ void gat_f32_to_f16(const float* __restrict__ in,
                               _Float16* __restrict__ out, int n4) {
  int t = blockIdx.x * blockDim.x + threadIdx.x;
  if (t >= n4) return;
  float4 v = *(const float4*)(in + (size_t)t * 4);
  _Float16* o = out + (size_t)t * 4;
  o[0] = (_Float16)v.x; o[1] = (_Float16)v.y;
  o[2] = (_Float16)v.z; o[3] = (_Float16)v.w;
}

// ---------------------------------------------------------------------------
// GEMM: C[nrows x D] = Af[nrows x 128] * W[128 x D]
// Af in f16 row-major, W pre-packed f16 fragments, C fp32.
// One wave32 per 16x16 tile; K loop = 4 x v_wmma_f32_16x16x32_f16.
// Branchless inner loop: 2 b128 loads (A) + 2 b128 loads (B) + 1 WMMA.
// OOB A rows are clamped (their products land only in unsaved C rows).
// ---------------------------------------------------------------------------
__global__ __launch_bounds__(128)
void gat_gemm_wmma(const _Float16* __restrict__ Af, const _Float16* __restrict__ Wp,
                   float* __restrict__ C, int nrows, int D) {
  const int lane = threadIdx.x;                                  // 0..31
  const int nt   = threadIdx.y + blockIdx.y * blockDim.y;        // N tile
  const int mt   = blockIdx.x;                                   // M tile
  const int n0 = nt * 16, m0 = mt * 16;
  if (n0 >= D || m0 >= nrows) return;                            // wave-uniform

  const bool hi = (lane >= 16);
  const int  r  = hi ? lane - 16 : lane;                         // 0..15
  int arow = m0 + r;
  if (arow >= nrows) arow = nrows - 1;                           // branchless clamp
  const _Float16* arp = Af + (size_t)arow * 128;                 // A row (f16)
  const _Float16* wfp = Wp + ((size_t)(nt * 4) * 32 + lane) * 16;
  const int kofs = hi ? 8 : 0;

  v8f acc = {};
  #pragma unroll
  for (int kk = 0; kk < 4; ++kk) {
    const int ka = kk * 32 + kofs;
    union { uint4 q[2]; v16h v; } au, bu;
    au.q[0] = *(const uint4*)(arp + ka);          // K = ka..ka+7
    au.q[1] = *(const uint4*)(arp + ka + 16);     // K = ka+16..ka+23
    bu.q[0] = *(const uint4*)(wfp + kk * 512);
    bu.q[1] = *(const uint4*)(wfp + kk * 512 + 8);
    acc = __builtin_amdgcn_wmma_f32_16x16x32_f16(
        /*neg_a=*/false, au.v, /*neg_b=*/false, bu.v,
        /*c_mod=*/(short)0, acc, /*reuse_a=*/false, /*reuse_b=*/false);
  }
  // C layout: VGPR i -> lanes 0-15: (M=i, N=lane); lanes 16-31: (M=8+i, N=lane-16)
  const int cm = m0 + (hi ? 8 : 0);
  const int cn = n0 + r;
  float* cp = C + (size_t)cm * D + cn;
  if (cm + 7 < nrows) {                       // fast path: full 8-row tile column
    #pragma unroll
    for (int i = 0; i < 8; ++i) cp[(size_t)i * D] = acc[i];
  } else {                                    // rare tail tile
    #pragma unroll
    for (int i = 0; i < 8; ++i)
      if (cm + i < nrows) cp[(size_t)i * D] = acc[i];
  }
}

// ---------------------------------------------------------------------------
// Per-node attention halves: as/ad[n*H+h] = dot(h[n,h,:], a_src/a_dst[h,:])
// ---------------------------------------------------------------------------
__global__ void gat_alpha(const float* __restrict__ h,
                          const float* __restrict__ a_src,
                          const float* __restrict__ a_dst,
                          float* __restrict__ as, float* __restrict__ ad,
                          int N, int H, int Cc) {
  int t = blockIdx.x * blockDim.x + threadIdx.x;
  if (t >= N * H) return;
  int n = t / H, hd = t - n * H;
  const float* hp = h + ((size_t)n * H + hd) * Cc;
  const float* sp = a_src + hd * Cc;
  const float* dp = a_dst + hd * Cc;
  float accs = 0.f, accd = 0.f;
  for (int c = 0; c < Cc; ++c) { float v = hp[c]; accs += v * sp[c]; accd += v * dp[c]; }
  as[t] = accs; ad[t] = accd;
}

// ---------------------------------------------------------------------------
// Init segment-max / denominator / aggregation buffers (every call: graph replay)
// ---------------------------------------------------------------------------
__global__ void gat_init(float* __restrict__ emax, float* __restrict__ denom,
                         float* __restrict__ agg, int NH, int ND) {
  int t = blockIdx.x * blockDim.x + threadIdx.x;
  if (t < NH) { emax[t] = -1e30f; denom[t] = 0.f; }
  if (t < ND) agg[t] = 0.f;
}

__device__ inline void atomicMaxF(float* addr, float val) {
  int* ai = (int*)addr;
  int cur = __float_as_int(*addr);
  while (__int_as_float(cur) < val) {
    int prev = atomicCAS(ai, cur, __float_as_int(val));
    if (prev == cur) break;
    cur = prev;
  }
}

// ---------------------------------------------------------------------------
// Pass 1: raw edge score (leaky-relu) + segment max over dst
// ---------------------------------------------------------------------------
__global__ void gat_edge_max(const int* __restrict__ src, const int* __restrict__ dst,
                             const float* __restrict__ as, const float* __restrict__ ad,
                             float* __restrict__ score, float* __restrict__ emax,
                             int E, int N, int H) {
  int t = blockIdx.x * blockDim.x + threadIdx.x;
  int total = (E + N) * H;
  if (t >= total) return;
  int e = t / H, hd = t - e * H;
  int s, d;
  if (e < E) { s = src[e]; d = dst[e]; } else { s = d = e - E; }  // self loops
  float v = as[s * H + hd] + ad[d * H + hd];
  v = (v > 0.f) ? v : NEG_SLOPE * v;
  score[t] = v;
  atomicMaxF(&emax[d * H + hd], v);
}

// ---------------------------------------------------------------------------
// Pass 2: ex = exp(score - max); segment sum over dst; score <- ex
// ---------------------------------------------------------------------------
__global__ void gat_edge_exp(const int* __restrict__ dst,
                             float* __restrict__ score,
                             const float* __restrict__ emax,
                             float* __restrict__ denom,
                             int E, int N, int H) {
  int t = blockIdx.x * blockDim.x + threadIdx.x;
  int total = (E + N) * H;
  if (t >= total) return;
  int e = t / H, hd = t - e * H;
  int d = (e < E) ? dst[e] : (e - E);
  float ex = __expf(score[t] - emax[d * H + hd]);
  score[t] = ex;
  atomicAdd(&denom[d * H + hd], ex);
}

// ---------------------------------------------------------------------------
// Pass 3: agg[dst] += h[src] * alpha.  One wave32 per edge, float4 per lane.
// ---------------------------------------------------------------------------
__global__ __launch_bounds__(256)
void gat_scatter(const int* __restrict__ src, const int* __restrict__ dst,
                 const float* __restrict__ score, const float* __restrict__ denom,
                 const float* __restrict__ h, float* __restrict__ agg,
                 int E, int N, int H, int D) {
  int lane = threadIdx.x & 31;
  int wid  = (blockIdx.x * blockDim.x + threadIdx.x) >> 5;
  if (wid >= E + N) return;
  int s, d;
  if (wid < E) { s = src[wid]; d = dst[wid]; } else { s = d = wid - E; }
  const int Cc = D / H;                      // channels per head (32,128,64: %4==0)
  for (int c = lane * 4; c < D; c += 128) {
    float4 hv = *(const float4*)(h + (size_t)s * D + c);
    int hd = c / Cc;
    float w = score[wid * H + hd] / (denom[d * H + hd] + 1e-16f);
    float* ap = agg + (size_t)d * D + c;
    atomicAdd(ap + 0, hv.x * w);
    atomicAdd(ap + 1, hv.y * w);
    atomicAdd(ap + 2, hv.z * w);
    atomicAdd(ap + 3, hv.w * w);
  }
}

// ---------------------------------------------------------------------------
// Finalize: relu(agg + bias).  f16 variant feeds the next layer's WMMA GEMM;
// f32 variant writes the final output. (Layer3 H=1 -> mean == identity.)
// ---------------------------------------------------------------------------
__global__ void gat_finalize_f16(const float* __restrict__ agg, const float* __restrict__ b,
                                 _Float16* __restrict__ out, int N, int D) {
  int t = blockIdx.x * blockDim.x + threadIdx.x;
  if (t >= N * D) return;
  float v = agg[t] + b[t % D];
  out[t] = (_Float16)(v > 0.f ? v : 0.f);
}

__global__ void gat_finalize_f32(const float* __restrict__ agg, const float* __restrict__ b,
                                 float* __restrict__ out, int N, int D) {
  int t = blockIdx.x * blockDim.x + threadIdx.x;
  if (t >= N * D) return;
  float v = agg[t] + b[t % D];
  out[t] = v > 0.f ? v : 0.f;
}

// ---------------------------------------------------------------------------
static void run_layer(hipStream_t st, const _Float16* inf16, const float* W,
                      const float* a_s, const float* a_d, const float* b,
                      _Float16* out16, float* out32,
                      int N, int E, const int* src, const int* dst,
                      int H, int Cc, float* hbuf, float* agg, _Float16* wpack,
                      float* score, float* emax, float* denom,
                      float* asb, float* adb) {
  const int D = H * Cc;
  const int nfragT = (D / 16) * 4 * 32;                // pack threads
  gat_pack_w<<<(nfragT + 255) / 256, 256, 0, st>>>(W, wpack, D);
  dim3 gb(32, 4);
  dim3 gg((N + 15) / 16, (D + 63) / 64);
  gat_gemm_wmma<<<gg, gb, 0, st>>>(inf16, wpack, hbuf, N, D);
  int NH = N * H;
  gat_alpha<<<(NH + 255) / 256, 256, 0, st>>>(hbuf, a_s, a_d, asb, adb, N, H, Cc);
  int ND = N * D;
  gat_init<<<(ND + 255) / 256, 256, 0, st>>>(emax, denom, agg, NH, ND);
  int TH = (E + N) * H;
  gat_edge_max<<<(TH + 255) / 256, 256, 0, st>>>(src, dst, asb, adb, score, emax, E, N, H);
  gat_edge_exp<<<(TH + 255) / 256, 256, 0, st>>>(dst, score, emax, denom, E, N, H);
  long long threads = (long long)(E + N) * 32;
  gat_scatter<<<(int)((threads + 255) / 256), 256, 0, st>>>(src, dst, score, denom,
                                                            hbuf, agg, E, N, H, D);
  if (out16)
    gat_finalize_f16<<<(ND + 255) / 256, 256, 0, st>>>(agg, b, out16, N, D);
  else
    gat_finalize_f32<<<(ND + 255) / 256, 256, 0, st>>>(agg, b, out32, N, D);
}

extern "C" void kernel_launch(void* const* d_in, const int* in_sizes, int n_in,
                              void* d_out, int out_size, void* d_ws, size_t ws_size,
                              hipStream_t stream) {
  const float* x     = (const float*)d_in[0];
  const int*   ei    = (const int*)d_in[1];
  const float* W1    = (const float*)d_in[2];
  const float* a1s   = (const float*)d_in[3];
  const float* a1d   = (const float*)d_in[4];
  const float* b1    = (const float*)d_in[5];
  const float* W2    = (const float*)d_in[6];
  const float* a2s   = (const float*)d_in[7];
  const float* a2d   = (const float*)d_in[8];
  const float* b2    = (const float*)d_in[9];
  const float* W3    = (const float*)d_in[10];
  const float* a3s   = (const float*)d_in[11];
  const float* a3d   = (const float*)d_in[12];
  const float* b3    = (const float*)d_in[13];

  const int N = in_sizes[0] / 128;
  const int E = in_sizes[1] / 2;
  const int* src = ei;
  const int* dst = ei + E;

  // Carve workspace (256B-aligned slabs)
  char* base = (char*)d_ws;
  size_t off = 0;
  auto carve = [&](size_t nbytes) {
    void* p = base + off;
    off += (nbytes + 255) & ~(size_t)255;
    return p;
  };
  float*    hbuf  = (float*)   carve((size_t)N * 128 * 4);   // transformed h (fp32)
  float*    agg   = (float*)   carve((size_t)N * 128 * 4);   // message accumulator
  _Float16* feath = (_Float16*)carve((size_t)N * 128 * 2);   // f16 features (GEMM A)
  _Float16* wpack = (_Float16*)carve((size_t)8 * 4 * 32 * 16 * 2); // packed W frags
  float*    score = (float*)   carve((size_t)(E + N) * 4 * 4);
  float*    emax  = (float*)   carve((size_t)N * 4 * 4);
  float*    denom = (float*)   carve((size_t)N * 4 * 4);
  float*    asb   = (float*)   carve((size_t)N * 4 * 4);
  float*    adb   = (float*)   carve((size_t)N * 4 * 4);
  (void)ws_size; (void)n_in; (void)out_size;

  // Convert layer-1 input to f16 once
  int n4 = (N * 128) / 4;
  gat_f32_to_f16<<<(n4 + 255) / 256, 256, 0, stream>>>(x, feath, n4);

  // Layer 1: 128 -> 4 heads x 32, concat  (f16 features out)
  run_layer(stream, feath, W1, a1s, a1d, b1, feath, nullptr, N, E, src, dst, 4, 32,
            hbuf, agg, wpack, score, emax, denom, asb, adb);
  // Layer 2: 128 -> 1 head x 128, concat  (f16 features out)
  run_layer(stream, feath, W2, a2s, a2d, b2, feath, nullptr, N, E, src, dst, 1, 128,
            hbuf, agg, wpack, score, emax, denom, asb, adb);
  // Layer 3: 128 -> 1 head x 64, mean (H=1 => identity), fp32 output
  run_layer(stream, feath, W3, a3s, a3d, b3, nullptr, (float*)d_out, N, E, src, dst, 1, 64,
            hbuf, agg, wpack, score, emax, denom, asb, adb);
}
